// SaTNC_46780783788234
// MI455X (gfx1250) — compile-verified, matrix operand order checked by reading, not run on hardware
//
#include <hip/hip_runtime.h>
#include <hip/hip_bf16.h>

// ---------------------------------------------------------------------------
// Problem constants (from the reference)
// ---------------------------------------------------------------------------
#define SEQ   17
#define DM    6
#define NH    4
#define DKH   4
#define DFH   16      // NH*DKH
#define DFF_  64
#define NL    2
#define SFAC  0.04f
#define EPSV  1e-5f
#define RPAD  128     // 108 padded to 128 (K of GEMM1)

typedef __attribute__((ext_vector_type(16))) _Float16 v16h;
typedef __attribute__((ext_vector_type(8)))  _Float16 h8;
typedef __attribute__((ext_vector_type(8)))  float    v8f;

// ---------------------------------------------------------------------------
// Utility kernels
// ---------------------------------------------------------------------------
__global__ void zero_kernel(float* __restrict__ p, int n) {
    int i = blockIdx.x * blockDim.x + threadIdx.x;
    if (i < n) p[i] = 0.0f;
}

// Build transposed, padded f16 weight matrices for WMMA B fragments:
//  B1t[n][k] (256 x 128) from lw1[k][n] (108 x 256), zero-padded k>=108
//  B2t[n][k] (512 x 256) from lw2[k][n] (256 x 512)
__global__ void prep_weights_kernel(const float* __restrict__ lw1,
                                    const float* __restrict__ lw2,
                                    _Float16* __restrict__ B1t,
                                    _Float16* __restrict__ B2t) {
    int t = blockIdx.x * blockDim.x + threadIdx.x;   // up to 512*256
    if (t < 256 * 128) {
        int n = t >> 7, k = t & 127;
        B1t[t] = (k < 108) ? (_Float16)lw1[k * 256 + n] : (_Float16)0.0f;
    }
    if (t < 512 * 256) {
        int n = t >> 8, k = t & 255;
        B2t[t] = (_Float16)lw2[k * 512 + n];
    }
}

// ---------------------------------------------------------------------------
// Transformer encoder: one thread per batch row, per-thread state in LDS
// ---------------------------------------------------------------------------
#define TPB_ENC 64

__device__ __forceinline__ void ln6(const float* y, const float* g, const float* bta,
                                    float* xo) {
    float m = 0.f;
#pragma unroll
    for (int d = 0; d < DM; d++) m += y[d];
    m *= (1.0f / DM);
    float v = 0.f;
#pragma unroll
    for (int d = 0; d < DM; d++) { float t = y[d] - m; v += t * t; }
    v *= (1.0f / DM);
    float r = rsqrtf(v + EPSV);
#pragma unroll
    for (int d = 0; d < DM; d++) xo[d] = g[d] * (y[d] - m) * r + bta[d];
}

__global__ __launch_bounds__(TPB_ENC) void encoder_kernel(
    const float* __restrict__ x0, const float* __restrict__ c,
    const int* __restrict__ element, const float* __restrict__ emb,
    const float* __restrict__ wq, const float* __restrict__ wk,
    const float* __restrict__ wv, const float* __restrict__ wo,
    const float* __restrict__ bo, const float* __restrict__ g1,
    const float* __restrict__ b1, const float* __restrict__ w1,
    const float* __restrict__ bf1, const float* __restrict__ w2,
    const float* __restrict__ bf2, const float* __restrict__ g2,
    const float* __restrict__ b2, _Float16* __restrict__ rh, int Btot) {
    __shared__ float xsh[SEQ * DM * TPB_ENC];        // x  [17][6]  per thread
    __shared__ float zsh[SEQ * DFH * TPB_ENC];       // z  [17][16] per thread
    __shared__ float kvsh[2 * SEQ * DKH * TPB_ENC];  // k,v[17][4]  per thread
    __shared__ float scsh[SEQ * TPB_ENC];            // scores [17] per thread

    const int tid = threadIdx.x;
    const int b = blockIdx.x * TPB_ENC + tid;
    if (b >= Btot) return;

#define XI(s, d)  xsh[((s) * DM + (d)) * TPB_ENC + tid]
#define ZI(s, f)  zsh[((s) * DFH + (f)) * TPB_ENC + tid]
#define KI(t, k)  kvsh[((t) * DKH + (k)) * TPB_ENC + tid]
#define VI(t, k)  kvsh[(SEQ * DKH + (t) * DKH + (k)) * TPB_ENC + tid]
#define SCI(t)    scsh[(t) * TPB_ENC + tid]

    const float* xb = x0 + (size_t)b * 2 * SEQ * DM;
    for (int s = 0; s < SEQ; s++) {
        int e = element[s];
        for (int d = 0; d < DM; d++) {
            float mask = xb[s * DM + d];
            float feat = xb[SEQ * DM + s * DM + d];
            XI(s, d) = mask * (feat + emb[e * DM + d] * SFAC);
        }
    }

    for (int l = 0; l < NL; l++) {
        const float* wql = wq + (size_t)l * NH * DM * DKH;
        const float* wkl = wk + (size_t)l * NH * DM * DKH;
        const float* wvl = wv + (size_t)l * NH * DM * DKH;

        for (int i = 0; i < SEQ * DFH; i++) zsh[i * TPB_ENC + tid] = 0.0f;

        for (int h = 0; h < NH; h++) {
            // K,V projections for this head
            for (int t = 0; t < SEQ; t++) {
#pragma unroll
                for (int kk = 0; kk < DKH; kk++) {
                    float kv = 0.f, vv = 0.f;
#pragma unroll
                    for (int d = 0; d < DM; d++) {
                        float xv = XI(t, d);
                        kv += xv * wkl[(h * DM + d) * DKH + kk];
                        vv += xv * wvl[(h * DM + d) * DKH + kk];
                    }
                    KI(t, kk) = kv;
                    VI(t, kk) = vv;
                }
            }
            // attention rows
            for (int s = 0; s < SEQ; s++) {
                float q[DKH];
#pragma unroll
                for (int kk = 0; kk < DKH; kk++) {
                    float qv = 0.f;
#pragma unroll
                    for (int d = 0; d < DM; d++)
                        qv += XI(s, d) * wql[(h * DM + d) * DKH + kk];
                    q[kk] = qv;
                }
                float mx = -1e30f;
                for (int t = 0; t < SEQ; t++) {
                    float dot = 0.f;
#pragma unroll
                    for (int kk = 0; kk < DKH; kk++) dot += q[kk] * KI(t, kk);
                    dot = (dot == 0.0f) ? -1e9f : dot;
                    dot *= (1.0f / SFAC);
                    SCI(t) = dot;
                    mx = fmaxf(mx, dot);
                }
                float ssum = 0.f;
                for (int t = 0; t < SEQ; t++) {
                    float e = __expf(SCI(t) - mx);
                    SCI(t) = e;
                    ssum += e;
                }
                float inv = 1.0f / ssum;
                for (int t = 0; t < SEQ; t++) {
                    float a = SCI(t) * inv;
#pragma unroll
                    for (int kk = 0; kk < DKH; kk++)
                        ZI(s, h * DKH + kk) += a * VI(t, kk);
                }
            }
        }

        // output projection + residual + LN1
        const float* wol = wo + (size_t)l * DFH * DM;
        const float* bol = bo + l * DM;
        const float* g1l = g1 + l * DM;
        const float* b1l = b1 + l * DM;
        for (int s = 0; s < SEQ; s++) {
            float att[DM];
#pragma unroll
            for (int d = 0; d < DM; d++) att[d] = bol[d];
            for (int f = 0; f < DFH; f++) {
                float zz = ZI(s, f);
#pragma unroll
                for (int d = 0; d < DM; d++) att[d] += zz * wol[f * DM + d];
            }
            float y[DM], xo[DM];
#pragma unroll
            for (int d = 0; d < DM; d++) y[d] = XI(s, d) + att[d];
            ln6(y, g1l, b1l, xo);
#pragma unroll
            for (int d = 0; d < DM; d++) XI(s, d) = xo[d];
        }

        // FFN + residual + LN2
        const float* w1l = w1 + (size_t)l * DM * DFF_;
        const float* bf1l = bf1 + l * DFF_;
        const float* w2l = w2 + (size_t)l * DFF_ * DM;
        const float* bf2l = bf2 + l * DM;
        const float* g2l = g2 + l * DM;
        const float* b2l = b2 + l * DM;
        for (int s = 0; s < SEQ; s++) {
            float f[DM];
#pragma unroll
            for (int d = 0; d < DM; d++) f[d] = bf2l[d];
            for (int j = 0; j < DFF_; j++) {
                float hj = bf1l[j];
#pragma unroll
                for (int d = 0; d < DM; d++) hj += XI(s, d) * w1l[d * DFF_ + j];
                hj = (hj > 0.f) ? hj : 0.01f * hj;     // leaky relu
#pragma unroll
                for (int d = 0; d < DM; d++) f[d] += hj * w2l[j * DM + d];
            }
            float y[DM], xo[DM];
#pragma unroll
            for (int d = 0; d < DM; d++) y[d] = XI(s, d) + f[d];
            ln6(y, g2l, b2l, xo);
#pragma unroll
            for (int d = 0; d < DM; d++) XI(s, d) = xo[d];
        }
    }

    // write r (f16, padded to 128): [x.flatten(102) | c(6) | zeros(20)]
    _Float16* out = rh + (size_t)b * RPAD;
    for (int s = 0; s < SEQ; s++)
        for (int d = 0; d < DM; d++)
            out[s * DM + d] = (_Float16)XI(s, d);
#pragma unroll
    for (int d = 0; d < DM; d++) out[SEQ * DM + d] = (_Float16)c[(size_t)b * DM + d];
#pragma unroll
    for (int i = SEQ * DM + DM; i < RPAD; i++) out[i] = (_Float16)0.0f;
#undef XI
#undef ZI
#undef KI
#undef VI
#undef SCI
}

// ---------------------------------------------------------------------------
// WMMA fragment loaders (16x16x32 f16; A/B rows striped per documented layout:
// lanes 0-15 hold K {kc*32 + 0..7, 16..23}, lanes 16-31 hold {8..15, 24..31})
// ---------------------------------------------------------------------------
__device__ __forceinline__ v16h load_frag_f16(const _Float16* __restrict__ base,
                                              int stride, int row, int kbase, int lane) {
    int khalf = (lane >> 4) << 3;  // 0 or 8
    const _Float16* p = base + (size_t)row * stride + kbase + khalf;
    h8 lo = *(const h8*)(p);
    h8 hi = *(const h8*)(p + 16);
    v16h a;
#pragma unroll
    for (int i = 0; i < 8; i++) { a[i] = lo[i]; a[i + 8] = hi[i]; }
    return a;
}

// A fragment from f32 activations with fused BN (scale/shift) + ReLU + f16 pack
__device__ __forceinline__ v16h load_frag_bnrelu(const float* __restrict__ base,
                                                 int stride, int row, int kbase, int lane,
                                                 const float* __restrict__ sc,
                                                 const float* __restrict__ sh) {
    int khalf = (lane >> 4) << 3;
    const float* p = base + (size_t)row * stride + kbase + khalf;
    const float* scp = sc + kbase + khalf;
    const float* shp = sh + kbase + khalf;
    v16h a;
#pragma unroll
    for (int i = 0; i < 8; i++) {
        float v0 = fmaxf(p[i] * scp[i] + shp[i], 0.0f);
        float v1 = fmaxf(p[16 + i] * scp[16 + i] + shp[16 + i], 0.0f);
        a[i] = (_Float16)v0;
        a[i + 8] = (_Float16)v1;
    }
    return a;
}

// ---------------------------------------------------------------------------
// GEMM1: y1[B,256] = rh[B,128] @ lw1[128,256] + lb1   (4 wmma per tile)
// Block = 4 waves; wave w owns M-tile (blockIdx.y*4+w), N-tile blockIdx.x.
// ---------------------------------------------------------------------------
__global__ __launch_bounds__(128) void gemm1_kernel(const _Float16* __restrict__ rh,
                                                    const _Float16* __restrict__ B1t,
                                                    const float* __restrict__ lb1,
                                                    float* __restrict__ y1) {
    int lane = threadIdx.x & 31;
    int wave = threadIdx.x >> 5;
    int mtile = blockIdx.y * 4 + wave;
    int mrow = mtile * 16 + (lane & 15);
    int ncol = blockIdx.x * 16 + (lane & 15);
    v8f acc = {0.f, 0.f, 0.f, 0.f, 0.f, 0.f, 0.f, 0.f};
#pragma unroll
    for (int kc = 0; kc < 4; kc++) {
        v16h a = load_frag_f16(rh, RPAD, mrow, kc * 32, lane);
        v16h bfr = load_frag_f16(B1t, RPAD, ncol, kc * 32, lane);
        acc = __builtin_amdgcn_wmma_f32_16x16x32_f16(false, a, false, bfr,
                                                     (short)0, acc, false, false);
    }
    float bias = lb1[ncol];
    int rbase = mtile * 16 + ((lane >> 4) << 3);
#pragma unroll
    for (int r = 0; r < 8; r++)
        y1[(size_t)(rbase + r) * 256 + ncol] = acc[r] + bias;
}

// ---------------------------------------------------------------------------
// GEMM2: y2[B,512] = relu(bn1(y1)) @ lw2[256,512] + lb2   (8 wmma per tile)
// ---------------------------------------------------------------------------
__global__ __launch_bounds__(128) void gemm2_kernel(const float* __restrict__ y1,
                                                    const _Float16* __restrict__ B2t,
                                                    const float* __restrict__ lb2,
                                                    const float* __restrict__ sc1,
                                                    const float* __restrict__ sh1,
                                                    float* __restrict__ y2) {
    int lane = threadIdx.x & 31;
    int wave = threadIdx.x >> 5;
    int mtile = blockIdx.y * 4 + wave;
    int mrow = mtile * 16 + (lane & 15);
    int ncol = blockIdx.x * 16 + (lane & 15);
    v8f acc = {0.f, 0.f, 0.f, 0.f, 0.f, 0.f, 0.f, 0.f};
#pragma unroll
    for (int kc = 0; kc < 8; kc++) {
        v16h a = load_frag_bnrelu(y1, 256, mrow, kc * 32, lane, sc1, sh1);
        v16h bfr = load_frag_f16(B2t, 256, ncol, kc * 32, lane);
        acc = __builtin_amdgcn_wmma_f32_16x16x32_f16(false, a, false, bfr,
                                                     (short)0, acc, false, false);
    }
    float bias = lb2[ncol];
    int rbase = mtile * 16 + ((lane >> 4) << 3);
#pragma unroll
    for (int r = 0; r < 8; r++)
        y2[(size_t)(rbase + r) * 512 + ncol] = acc[r] + bias;
}

// ---------------------------------------------------------------------------
// Per-column sum / sum-of-squares (BatchNorm batch statistics)
// ---------------------------------------------------------------------------
__global__ __launch_bounds__(256) void colstats_kernel(const float* __restrict__ y,
                                                       int N, int rows_per_block,
                                                       float* __restrict__ sum,
                                                       float* __restrict__ sumsq) {
    int col = blockIdx.x * 256 + threadIdx.x;
    size_t r0 = (size_t)blockIdx.y * rows_per_block;
    float s = 0.f, q = 0.f;
    for (int i = 0; i < rows_per_block; i++) {
        float v = y[(r0 + i) * N + col];
        s += v;
        q += v * v;
    }
    atomicAdd(&sum[col], s);
    atomicAdd(&sumsq[col], q);
}

// Fold mean/var/gamma/beta into per-feature scale & shift.
__global__ void finalize_bn_kernel(const float* __restrict__ sum,
                                   const float* __restrict__ sumsq,
                                   const float* __restrict__ g,
                                   const float* __restrict__ bta,
                                   float* __restrict__ sc, float* __restrict__ sh,
                                   int N, float invB) {
    int i = blockIdx.x * blockDim.x + threadIdx.x;
    if (i < N) {
        float m = sum[i] * invB;
        float v = sumsq[i] * invB - m * m;
        float r = rsqrtf(v + EPSV);
        float scale = g[i] * r;
        sc[i] = scale;
        sh[i] = bta[i] - m * scale;
    }
}

// ---------------------------------------------------------------------------
// Final: out[b] = relu(bn2(y2[b,:])) . fw + fb    (one wave per row)
// ---------------------------------------------------------------------------
__global__ __launch_bounds__(256) void head_final_kernel(const float* __restrict__ y2,
                                                         const float* __restrict__ sc2,
                                                         const float* __restrict__ sh2,
                                                         const float* __restrict__ fw,
                                                         const float* __restrict__ fb,
                                                         float* __restrict__ out) {
    int wave = threadIdx.x >> 5;
    int lane = threadIdx.x & 31;
    int b = blockIdx.x * 8 + wave;
    const float* row = y2 + (size_t)b * 512;
    float acc = 0.f;
#pragma unroll
    for (int i = 0; i < 16; i++) {
        int j = lane + i * 32;
        float v = fmaxf(row[j] * sc2[j] + sh2[j], 0.0f);
        acc += v * fw[j];
    }
#pragma unroll
    for (int off = 16; off > 0; off >>= 1) acc += __shfl_xor(acc, off, 32);
    if (lane == 0) out[b] = acc + fb[0];
}

// ---------------------------------------------------------------------------
// Launcher
// ---------------------------------------------------------------------------
extern "C" void kernel_launch(void* const* d_in, const int* in_sizes, int n_in,
                              void* d_out, int out_size, void* d_ws, size_t ws_size,
                              hipStream_t stream) {
    const float* x0  = (const float*)d_in[0];
    const float* c   = (const float*)d_in[1];
    const int* element = (const int*)d_in[2];
    const float* emb = (const float*)d_in[3];
    const float* wq  = (const float*)d_in[4];
    const float* wk  = (const float*)d_in[5];
    const float* wv  = (const float*)d_in[6];
    const float* wo  = (const float*)d_in[7];
    const float* bo  = (const float*)d_in[8];
    const float* g1  = (const float*)d_in[9];
    const float* b1  = (const float*)d_in[10];
    const float* w1  = (const float*)d_in[11];
    const float* bf1 = (const float*)d_in[12];
    const float* w2  = (const float*)d_in[13];
    const float* bf2 = (const float*)d_in[14];
    const float* g2  = (const float*)d_in[15];
    const float* b2  = (const float*)d_in[16];
    const float* lw1 = (const float*)d_in[17];
    const float* lb1 = (const float*)d_in[18];
    const float* bn1g = (const float*)d_in[19];
    const float* bn1b = (const float*)d_in[20];
    const float* lw2 = (const float*)d_in[21];
    const float* lb2 = (const float*)d_in[22];
    const float* bn2g = (const float*)d_in[23];
    const float* bn2b = (const float*)d_in[24];
    const float* fw  = (const float*)d_in[25];
    const float* fb  = (const float*)d_in[26];

    const int B = in_sizes[1] / DM;   // c is [B, D]

    // Workspace carve (all chunks 256B aligned by construction)
    char* ws = (char*)d_ws;
    float* y1 = (float*)ws;            ws += (size_t)B * 256 * sizeof(float);
    float* y2 = (float*)ws;            ws += (size_t)B * 512 * sizeof(float);
    _Float16* rh = (_Float16*)ws;      ws += (size_t)B * RPAD * sizeof(_Float16);
    _Float16* B1t = (_Float16*)ws;     ws += (size_t)256 * 128 * sizeof(_Float16);
    _Float16* B2t = (_Float16*)ws;     ws += (size_t)512 * 256 * sizeof(_Float16);
    float* stats = (float*)ws;         ws += 1536 * sizeof(float); // sum1,sq1,sum2,sq2
    float* sc1 = (float*)ws;           ws += 256 * sizeof(float);
    float* sh1 = (float*)ws;           ws += 256 * sizeof(float);
    float* sc2 = (float*)ws;           ws += 512 * sizeof(float);
    float* sh2 = (float*)ws;           ws += 512 * sizeof(float);

    float* sum1 = stats;
    float* sq1  = stats + 256;
    float* sum2 = stats + 512;
    float* sq2  = stats + 1024;

    // 1) zero BN stat accumulators
    zero_kernel<<<6, 256, 0, stream>>>(stats, 1536);

    // 2) transpose/pad weights to f16 [N][K]
    prep_weights_kernel<<<512, 256, 0, stream>>>(lw1, lw2, B1t, B2t);

    // 3) transformer encoder -> rh [B,128] f16
    encoder_kernel<<<B / TPB_ENC, TPB_ENC, 0, stream>>>(
        x0, c, element, emb, wq, wk, wv, wo, bo, g1, b1, w1, bf1, w2, bf2, g2, b2,
        rh, B);

    // 4) GEMM1 (WMMA): y1 = rh @ lw1 + lb1
    gemm1_kernel<<<dim3(16, B / 64), 128, 0, stream>>>(rh, B1t, lb1, y1);

    // 5) BN1 stats + finalize
    colstats_kernel<<<dim3(1, 128), 256, 0, stream>>>(y1, 256, B / 128, sum1, sq1);
    finalize_bn_kernel<<<1, 256, 0, stream>>>(sum1, sq1, bn1g, bn1b, sc1, sh1, 256,
                                              1.0f / (float)B);

    // 6) GEMM2 (WMMA, fused BN1+ReLU on A): y2 = relu(bn1(y1)) @ lw2 + lb2
    gemm2_kernel<<<dim3(32, B / 64), 128, 0, stream>>>(y1, B2t, lb2, sc1, sh1, y2);

    // 7) BN2 stats + finalize
    colstats_kernel<<<dim3(2, 128), 256, 0, stream>>>(y2, 512, B / 128, sum2, sq2);
    finalize_bn_kernel<<<2, 256, 0, stream>>>(sum2, sq2, bn2g, bn2b, sc2, sh2, 512,
                                              1.0f / (float)B);

    // 8) final dot with fw
    head_final_kernel<<<B / 8, 256, 0, stream>>>(y2, sc2, sh2, fw, fb, (float*)d_out);
}
